// SelfAttention_65481071402676
// MI455X (gfx1250) — matrix operand order repaired and checked by reading, not compiled, not run on hardware
//
#include <hip/hip_runtime.h>
#include <hip/hip_bf16.h>
#include <stdint.h>

typedef __bf16 bf16_t;
typedef bf16_t bf16x8  __attribute__((ext_vector_type(8)));
typedef bf16_t bf16x16 __attribute__((ext_vector_type(16)));
typedef float  f32x8   __attribute__((ext_vector_type(8)));
typedef unsigned int u32x4 __attribute__((ext_vector_type(4)));
typedef int i32x4 __attribute__((ext_vector_type(4)));
typedef int i32x8 __attribute__((ext_vector_type(8)));

#if __has_builtin(__builtin_amdgcn_tensor_load_to_lds) && __has_builtin(__builtin_amdgcn_s_wait_tensorcnt)
#define USE_TDM 1
#else
#define USE_TDM 0
#endif
#if __has_builtin(__builtin_amdgcn_mov_dpp)
#define HAVE_DPP 1
#else
#define HAVE_DPP 0
#endif

union FragAB {
    bf16x16 v;
    bf16x8  h[2];
};

__device__ __forceinline__ unsigned short f2bf(float f) {
    unsigned u = __float_as_uint(f);
    u += 0x7fffu + ((u >> 16) & 1u);   // round-to-nearest-even
    return (unsigned short)(u >> 16);
}

// all-reduce across the 16-lane column group (xor 1,2,4,8) on the VALU via DPP
__device__ __forceinline__ float xor16_max(float v) {
#if HAVE_DPP
    v = fmaxf(v, __int_as_float(__builtin_amdgcn_mov_dpp(__float_as_int(v), 0xB1,  0xF, 0xF, true))); // quad_perm xor1
    v = fmaxf(v, __int_as_float(__builtin_amdgcn_mov_dpp(__float_as_int(v), 0x4E,  0xF, 0xF, true))); // quad_perm xor2
    v = fmaxf(v, __int_as_float(__builtin_amdgcn_mov_dpp(__float_as_int(v), 0x141, 0xF, 0xF, true))); // row_half_mirror
    v = fmaxf(v, __int_as_float(__builtin_amdgcn_mov_dpp(__float_as_int(v), 0x140, 0xF, 0xF, true))); // row_mirror
#else
    v = fmaxf(v, __shfl_xor(v, 1));
    v = fmaxf(v, __shfl_xor(v, 2));
    v = fmaxf(v, __shfl_xor(v, 4));
    v = fmaxf(v, __shfl_xor(v, 8));
#endif
    return v;
}
__device__ __forceinline__ float xor16_add(float v) {
#if HAVE_DPP
    v += __int_as_float(__builtin_amdgcn_mov_dpp(__float_as_int(v), 0xB1,  0xF, 0xF, true));
    v += __int_as_float(__builtin_amdgcn_mov_dpp(__float_as_int(v), 0x4E,  0xF, 0xF, true));
    v += __int_as_float(__builtin_amdgcn_mov_dpp(__float_as_int(v), 0x141, 0xF, 0xF, true));
    v += __int_as_float(__builtin_amdgcn_mov_dpp(__float_as_int(v), 0x140, 0xF, 0xF, true));
#else
    v += __shfl_xor(v, 1);
    v += __shfl_xor(v, 2);
    v += __shfl_xor(v, 4);
    v += __shfl_xor(v, 8);
#endif
    return v;
}

#if USE_TDM
// TDM: DMA a [rows x 64] bf16 tile (row stride row_stride elems in memory) into
// LDS with a 72-element padded row stride (pad 4 DWORDs after every 32 DWORDs).
// This toolchain's builtin arity: (u32x4 g0, i32x8 g1, i32x4 g2, i32x4 g3, i32x8, i32 cpol)
__device__ __forceinline__ void tdm_load_tile_bf16(const unsigned short* gsrc,
                                                   const unsigned short* lds_dst,
                                                   int rows, int row_stride) {
    unsigned long long ga = (unsigned long long)(uintptr_t)gsrc;
    u32x4 g0;
    g0[0] = 1u;                                                  // count=1, user mode
    g0[1] = (unsigned)(uintptr_t)lds_dst;                        // lds_addr
    g0[2] = (unsigned)ga;                                        // global_addr[31:0]
    g0[3] = (unsigned)((ga >> 32) & 0x01ffffffu) | (2u << 30);   // global_addr[56:32] | type=2
    i32x8 g1;
    g1[0] = (int)((1u << 16)    // data_size = 2 bytes
                | (1u << 20)    // pad_enable
                | (4u << 22)    // pad_interval: every 32 DWORDs (= 64 bf16 row)
                | (3u << 25));  // pad_amount: 4 DWORDs (= 8 bf16)
    g1[1] = (int)(64u << 16);                   // tensor_dim0 = 64
    g1[2] = (int)((unsigned)rows << 16);        // tensor_dim1 = rows
    g1[3] = (int)(64u << 16);                   // tile_dim0 = 64
    g1[4] = rows;                               // tile_dim1 = rows
    g1[5] = row_stride;                         // tensor_dim0_stride (elements)
    g1[6] = 0;
    g1[7] = 0;
    i32x4 z4; z4[0] = 0; z4[1] = 0; z4[2] = 0; z4[3] = 0;
    i32x8 z8;
    #pragma unroll
    for (int i = 0; i < 8; ++i) z8[i] = 0;
    __builtin_amdgcn_tensor_load_to_lds(g0, g1, z4, z4, z8, 0);
}
#endif

// cooperative fallback copy: [rows x 64] -> LDS rows of stride 72 (2 threads/row)
__device__ __forceinline__ void copy_tile_64(const unsigned short* src,
                                             unsigned short (*dst)[72],
                                             int row_stride, int tid) {
    const int lrow = tid >> 1;
    const int lseg = (tid & 1) * 32;
    const uint4* p = (const uint4*)(src + (size_t)lrow * row_stride + lseg);
    #pragma unroll
    for (int i = 0; i < 4; ++i)
        *(uint4*)&dst[lrow][lseg + 8 * i] = p[i];
}

// ---------------------------------------------------------------------------
// f32 -> bf16 elementwise (vectorized x4)
// ---------------------------------------------------------------------------
__global__ __launch_bounds__(256)
void cvt_f32_bf16(const float* __restrict__ src, unsigned short* __restrict__ dst, int n4) {
    int i = blockIdx.x * 256 + threadIdx.x;
    if (i < n4) {
        float4 f = ((const float4*)src)[i];
        ushort4 o;
        o.x = f2bf(f.x); o.y = f2bf(f.y); o.z = f2bf(f.z); o.w = f2bf(f.w);
        ((ushort4*)dst)[i] = o;
    }
}

// ---------------------------------------------------------------------------
// C[M,N] = A[M,K] * B[N,K]^T   (bf16 in, nn.Linear layout: both K-contiguous)
// block = 128x128, 256 thr = 8 waves (2x4), wave = 64x32 (8 f32x8 accums).
// K step 64, TDM double-buffered LDS tiles (DMA of k+1 overlaps WMMA on k).
// ---------------------------------------------------------------------------
template<bool STORE_BF16>
__global__ __launch_bounds__(256)
void gemm_bf16(const unsigned short* __restrict__ A,
               const unsigned short* __restrict__ B,
               void* __restrict__ Cout,
               const float* __restrict__ bias,
               int M, int N, int K) {
    __shared__ unsigned short As[2][128][72];
    __shared__ unsigned short Bs[2][128][72];

    const int tid  = threadIdx.x;
    const int lane = tid & 31;
    const int wid  = tid >> 5;
    const int wm   = (wid >> 2) * 64;
    const int wn   = (wid & 3) * 32;
    const int bm   = blockIdx.x * 128;
    const int bn   = blockIdx.y * 128;
    const int m16  = lane & 15;
    const int hi   = lane >> 4;

    f32x8 acc[4][2];
    #pragma unroll
    for (int i = 0; i < 4; ++i) {
        #pragma unroll
        for (int j = 0; j < 2; ++j) {
            #pragma unroll
            for (int e = 0; e < 8; ++e) acc[i][j][e] = 0.0f;
        }
    }

    const unsigned short* aTile = A + (size_t)bm * K;
    const unsigned short* bTile = B + (size_t)bn * K;
    const int nk = K >> 6;

#if USE_TDM
    if (wid == 0) {
        tdm_load_tile_bf16(aTile, &As[0][0][0], 128, K);
        tdm_load_tile_bf16(bTile, &Bs[0][0][0], 128, K);
    }
#else
    copy_tile_64(aTile, As[0], K, tid);
    copy_tile_64(bTile, Bs[0], K, tid);
#endif

    for (int kk = 0; kk < nk; ++kk) {
        const int buf = kk & 1;
#if USE_TDM
        if (wid == 0) {
            if (kk + 1 < nk) {
                tdm_load_tile_bf16(aTile + (kk + 1) * 64, &As[buf ^ 1][0][0], 128, K);
                tdm_load_tile_bf16(bTile + (kk + 1) * 64, &Bs[buf ^ 1][0][0], 128, K);
                __builtin_amdgcn_s_wait_tensorcnt(2);   // current pair done, next in flight
            } else {
                __builtin_amdgcn_s_wait_tensorcnt(0);
            }
        }
#endif
        __syncthreads();

        #pragma unroll
        for (int s = 0; s < 2; ++s) {
            FragAB bfrag[2];
            #pragma unroll
            for (int tn = 0; tn < 2; ++tn) {
                const int col = wn + tn * 16 + m16;
                bfrag[tn].h[0] = *(const bf16x8*)&Bs[buf][col][s * 32 + hi * 16];
                bfrag[tn].h[1] = *(const bf16x8*)&Bs[buf][col][s * 32 + hi * 16 + 8];
            }
            #pragma unroll
            for (int tm = 0; tm < 4; ++tm) {
                const int row = wm + tm * 16 + m16;
                FragAB af;
                af.h[0] = *(const bf16x8*)&As[buf][row][s * 32 + hi * 16];
                af.h[1] = *(const bf16x8*)&As[buf][row][s * 32 + hi * 16 + 8];
                #pragma unroll
                for (int tn = 0; tn < 2; ++tn) {
                    acc[tm][tn] = __builtin_amdgcn_wmma_f32_16x16x32_bf16(
                        false, af.v, false, bfrag[tn].v, (short)0, acc[tm][tn], false, false);
                }
            }
        }
        __syncthreads();

#if !USE_TDM
        if (kk + 1 < nk) {
            copy_tile_64(aTile + (kk + 1) * 64, As[buf ^ 1], K, tid);
            copy_tile_64(bTile + (kk + 1) * 64, Bs[buf ^ 1], K, tid);
        }
#endif
    }

    // Epilogue: lane holds col = m16 (+16*tn), rows hi*8 + r (+16*tm)
    const int mb = hi * 8;
    #pragma unroll
    for (int tm = 0; tm < 4; ++tm) {
        #pragma unroll
        for (int tn = 0; tn < 2; ++tn) {
            const int col = bn + wn + tn * 16 + m16;
            float bv = 0.0f;
            if (!STORE_BF16) bv = bias[col];
            #pragma unroll
            for (int r = 0; r < 8; ++r) {
                const int rowg = bm + wm + tm * 16 + mb + r;
                float v = acc[tm][tn][r];
                if (STORE_BF16) {
                    ((unsigned short*)Cout)[(size_t)rowg * N + col] = f2bf(v);
                } else {
                    ((float*)Cout)[(size_t)rowg * N + col] = v + bv;
                }
            }
        }
    }
}

// ---------------------------------------------------------------------------
// Flash attention. qkv bf16 [B*N][3E]. grid.x = b*H+h, grid.y = 64-query tile.
// 128 thr = 4 waves, each wave owns 16 query rows. K tile via TDM, V staged
// transposed by all threads, online softmax with DPP reductions.
// ---------------------------------------------------------------------------
__global__ __launch_bounds__(128)
void attention_fwd(const unsigned short* __restrict__ qkv,
                   unsigned short* __restrict__ out) {
    constexpr int Nseq = 2048, E3 = 3072, E = 1024, Dh = 64;
    const float scale = 0.125f;   // 1/sqrt(64)

    __shared__ unsigned short Ks[64][72];      // [key][d]
    __shared__ unsigned short Vt[64][72];      // [d][key] (transposed)
    __shared__ unsigned short Ps[4][16][72];   // per-wave P staging [qrow][key]

    const int tid  = threadIdx.x;
    const int lane = tid & 31;
    const int wid  = tid >> 5;
    const int b    = blockIdx.x >> 4;
    const int h    = blockIdx.x & 15;
    const int q0   = blockIdx.y * 64 + wid * 16;
    const int m16  = lane & 15;
    const int hi   = lane >> 4;
    const int mb   = hi * 8;

    // Q fragments for this wave's 16 rows, kept in registers for whole kernel
    const unsigned short* qptr = qkv + (size_t)(b * Nseq + q0 + m16) * E3 + h * Dh;
    FragAB qa[2];
    #pragma unroll
    for (int s = 0; s < 2; ++s) {
        qa[s].h[0] = *(const bf16x8*)(qptr + s * 32 + hi * 16);
        qa[s].h[1] = *(const bf16x8*)(qptr + s * 32 + hi * 16 + 8);
    }

    float mrun[8], lrun[8];
    f32x8 o[4];
    #pragma unroll
    for (int r = 0; r < 8; ++r) { mrun[r] = -1e30f; lrun[r] = 0.0f; }
    #pragma unroll
    for (int t = 0; t < 4; ++t) {
        #pragma unroll
        for (int r = 0; r < 8; ++r) o[t][r] = 0.0f;
    }

    const int ldrow = tid >> 1;
    const int ldseg = (tid & 1) * 32;
    const unsigned short* kTile0   = qkv + (size_t)(b * Nseq) * E3 +     E + h * Dh;
    const unsigned short* vSrcBase = qkv + (size_t)(b * Nseq) * E3 + 2 * E + h * Dh + ldseg;

    for (int j0 = 0; j0 < Nseq; j0 += 64) {
        // ---- K chunk via TDM DMA; V staged transposed by all threads ----
#if USE_TDM
        if (wid == 0)
            tdm_load_tile_bf16(kTile0 + (size_t)j0 * E3, &Ks[0][0], 64, E3);
#else
        copy_tile_64(kTile0 + (size_t)j0 * E3, Ks, E3, tid);
#endif
        {
            union { uint4 u[4]; unsigned short s[32]; } vv;
            const uint4* vp = (const uint4*)(vSrcBase + (size_t)(j0 + ldrow) * E3);
            #pragma unroll
            for (int i = 0; i < 4; ++i) vv.u[i] = vp[i];
            #pragma unroll
            for (int e = 0; e < 32; ++e)
                Vt[ldseg + e][ldrow] = vv.s[e];
        }
#if USE_TDM
        if (wid == 0) __builtin_amdgcn_s_wait_tensorcnt(0);
#endif
        __syncthreads();

        // ---- S = scale * Q K^T (16 x 64 per wave) ----
        f32x8 S[4];
        #pragma unroll
        for (int t = 0; t < 4; ++t) {
            const int kr = t * 16 + m16;
            FragAB kb0, kb1;
            kb0.h[0] = *(const bf16x8*)&Ks[kr][hi * 16];
            kb0.h[1] = *(const bf16x8*)&Ks[kr][hi * 16 + 8];
            kb1.h[0] = *(const bf16x8*)&Ks[kr][32 + hi * 16];
            kb1.h[1] = *(const bf16x8*)&Ks[kr][32 + hi * 16 + 8];
            f32x8 sacc;
            #pragma unroll
            for (int r = 0; r < 8; ++r) sacc[r] = 0.0f;
            sacc = __builtin_amdgcn_wmma_f32_16x16x32_bf16(false, qa[0].v, false, kb0.v, (short)0, sacc, false, false);
            sacc = __builtin_amdgcn_wmma_f32_16x16x32_bf16(false, qa[1].v, false, kb1.v, (short)0, sacc, false, false);
            #pragma unroll
            for (int r = 0; r < 8; ++r) S[t][r] = sacc[r] * scale;
        }

        // ---- online softmax, row stats via DPP all-reduce ----
        #pragma unroll
        for (int r = 0; r < 8; ++r) {
            float v = fmaxf(fmaxf(S[0][r], S[1][r]), fmaxf(S[2][r], S[3][r]));
            v = xor16_max(v);
            const float mnew  = fmaxf(mrun[r], v);
            const float alpha = __expf(mrun[r] - mnew);
            float sum = 0.0f;
            #pragma unroll
            for (int t = 0; t < 4; ++t) {
                float p = __expf(S[t][r] - mnew);
                S[t][r] = p;
                sum += p;
            }
            sum = xor16_add(sum);
            lrun[r] = lrun[r] * alpha + sum;
            mrun[r] = mnew;
            #pragma unroll
            for (int t = 0; t < 4; ++t) o[t][r] *= alpha;
        }

        // ---- stage P through per-wave LDS to re-fragment as A operand ----
        #pragma unroll
        for (int t = 0; t < 4; ++t) {
            #pragma unroll
            for (int r = 0; r < 8; ++r)
                Ps[wid][mb + r][t * 16 + m16] = f2bf(S[t][r]);
        }
        // intra-wave DS ops are in-order: stores above complete before loads below

        // ---- O += P @ V ----
        #pragma unroll
        for (int s = 0; s < 2; ++s) {
            FragAB pf;
            pf.h[0] = *(const bf16x8*)&Ps[wid][m16][s * 32 + hi * 16];
            pf.h[1] = *(const bf16x8*)&Ps[wid][m16][s * 32 + hi * 16 + 8];
            #pragma unroll
            for (int t = 0; t < 4; ++t) {
                FragAB vf;
                vf.h[0] = *(const bf16x8*)&Vt[t * 16 + m16][s * 32 + hi * 16];
                vf.h[1] = *(const bf16x8*)&Vt[t * 16 + m16][s * 32 + hi * 16 + 8];
                o[t] = __builtin_amdgcn_wmma_f32_16x16x32_bf16(
                    false, pf.v, false, vf.v, (short)0, o[t], false, false);
            }
        }
        __syncthreads();
    }

    // ---- normalize and write bf16 [B*N][E] ----
    #pragma unroll
    for (int r = 0; r < 8; ++r) {
        const float inv = 1.0f / lrun[r];
        const size_t rowoff = (size_t)(b * Nseq + q0 + mb + r) * E + h * Dh;
        #pragma unroll
        for (int t = 0; t < 4; ++t)
            out[rowoff + t * 16 + m16] = f2bf(o[t][r] * inv);
    }
}

// ---------------------------------------------------------------------------
extern "C" void kernel_launch(void* const* d_in, const int* in_sizes, int n_in,
                              void* d_out, int out_size, void* d_ws, size_t ws_size,
                              hipStream_t stream) {
    const float* x    = (const float*)d_in[0];   // [4,2048,1024]
    const float* Wqkv = (const float*)d_in[1];   // [3072,1024]
    const float* Wout = (const float*)d_in[2];   // [1024,1024]
    const float* bout = (const float*)d_in[3];   // [1024]

    char* ws = (char*)d_ws;
    unsigned short* x_bf    = (unsigned short*)(ws);                       // 16 MB
    unsigned short* wqkv_bf = (unsigned short*)(ws + (size_t)16777216);    //  6 MB
    unsigned short* wout_bf = (unsigned short*)(ws + (size_t)23068672);    //  2 MB
    unsigned short* qkv_bf  = (unsigned short*)(ws + (size_t)25165824);    // 48 MB
    unsigned short* attn_bf = (unsigned short*)(ws + (size_t)75497472);    // 16 MB

    // 1) convert inputs to bf16
    cvt_f32_bf16<<<dim3(8388608 / 4 / 256), 256, 0, stream>>>(x,    x_bf,    8388608 / 4);
    cvt_f32_bf16<<<dim3(3145728 / 4 / 256), 256, 0, stream>>>(Wqkv, wqkv_bf, 3145728 / 4);
    cvt_f32_bf16<<<dim3(1048576 / 4 / 256), 256, 0, stream>>>(Wout, wout_bf, 1048576 / 4);

    // 2) QKV projection: [8192,3072] = x[8192,1024] @ Wqkv[3072,1024]^T
    gemm_bf16<true><<<dim3(64, 24), 256, 0, stream>>>(x_bf, wqkv_bf, qkv_bf, nullptr,
                                                      8192, 3072, 1024);

    // 3) flash attention per (b,h), 64-query tiles
    attention_fwd<<<dim3(64, 32), 128, 0, stream>>>(qkv_bf, attn_bf);

    // 4) output projection + bias into f32 d_out
    gemm_bf16<false><<<dim3(64, 8), 256, 0, stream>>>(attn_bf, wout_bf, d_out, bout,
                                                      8192, 1024, 1024);
}
